// CWST3_86509231276258
// MI455X (gfx1250) — matrix operand order repaired and checked by reading, not compiled, not run on hardware
//
#include <hip/hip_runtime.h>
#include <cmath>

typedef __attribute__((ext_vector_type(16))) _Float16 v16h;
typedef __attribute__((ext_vector_type(8)))  float    v8f;
typedef __attribute__((ext_vector_type(4)))  float    v4f;

constexpr int BATCH = 2;
constexpr int C_IN  = 64;
constexpr int SZ    = 320;
constexpr int HS    = 160;           // SZ/2
constexpr int CB    = 32;            // channels per half
constexpr int HEADS = 2;
constexpr int HD    = 16;            // CB / HEADS
constexpr int PD    = 8;             // dpb hidden
constexpr int MLPD  = 128;           // 4*CB
constexpr int LTOK  = SZ * SZ;       // 102400 tokens per batch
constexpr int MTOK  = BATCH * LTOK;  // 204800 GEMM rows

static inline int cdiv(int a, int b) { return (a + b - 1) / b; }

// ---------------------------------------------------------------------------
// DWT + feat re-pack:  x (B,64,320,320) -> f1/f2 as (B*L, 32)
// ---------------------------------------------------------------------------
static __device__ __forceinline__
void store_feat(float* __restrict__ f1, float* __restrict__ f2,
                int b, int ch, int h, int w, float v) {
  size_t flat = (size_t)ch * (SZ * SZ) + (size_t)h * SZ + w;
  size_t l = flat >> 6;            // / 64
  int    c = (int)(flat & 63);
  size_t row = (size_t)b * LTOK + l;
  if (c < CB) f1[row * CB + c] = v;
  else        f2[row * CB + (c - CB)] = v;
}

__global__ void dwt_feat_kernel(const float* __restrict__ x,
                                float* __restrict__ f1, float* __restrict__ f2) {
  int idx = blockIdx.x * blockDim.x + threadIdx.x;
  int total = BATCH * C_IN * HS * HS;
  if (idx >= total) return;
  int j = idx % HS;  int t = idx / HS;
  int i = t % HS;    t /= HS;
  int ch = t % C_IN; int b = t / C_IN;
  const float* xp = x + (((size_t)b * C_IN + ch) * SZ + 2 * i) * SZ + 2 * j;
  float a = xp[0], bb = xp[1], c = xp[SZ], d = xp[SZ + 1];
  float ll = 0.5f * ( a + bb + c + d);
  float lh = 0.5f * (-a - bb + c + d);
  float hl = 0.5f * (-a + bb - c + d);
  float hh = 0.5f * ( a - bb - c + d);
  store_feat(f1, f2, b, ch, i,      j,      ll);
  store_feat(f1, f2, b, ch, i,      j + HS, lh);
  store_feat(f1, f2, b, ch, i + HS, j,      hl);
  store_feat(f1, f2, b, ch, i + HS, j + HS, hh);
}

// ---------------------------------------------------------------------------
// LayerNorm over C=32 per token (vectorized b128 loads/stores)
// ---------------------------------------------------------------------------
__global__ void ln_kernel(const float* __restrict__ x, const float* __restrict__ g,
                          const float* __restrict__ b, float* __restrict__ out, int M) {
  int t = blockIdx.x * blockDim.x + threadIdx.x;
  if (t >= M) return;
  const v4f* r4 = reinterpret_cast<const v4f*>(x + (size_t)t * CB);
  v4f buf[CB / 4];
  #pragma unroll
  for (int i = 0; i < CB / 4; ++i) buf[i] = r4[i];
  float mu = 0.f;
  #pragma unroll
  for (int i = 0; i < CB / 4; ++i) mu += buf[i][0] + buf[i][1] + buf[i][2] + buf[i][3];
  mu *= (1.0f / CB);
  float var = 0.f;
  #pragma unroll
  for (int i = 0; i < CB / 4; ++i) {
    #pragma unroll
    for (int e = 0; e < 4; ++e) { float d = buf[i][e] - mu; var += d * d; }
  }
  var *= (1.0f / CB);
  float inv = rsqrtf(var + 1e-5f);
  v4f* o4 = reinterpret_cast<v4f*>(out + (size_t)t * CB);
  #pragma unroll
  for (int i = 0; i < CB / 4; ++i) {
    v4f o;
    #pragma unroll
    for (int e = 0; e < 4; ++e)
      o[e] = (buf[i][e] - mu) * inv * g[i * 4 + e] + b[i * 4 + e];
    o4[i] = o;
  }
}

// ---------------------------------------------------------------------------
// WMMA GEMM:  out(MxN) = A(MxK) @ W^T  (+bias) (+residual) (+GELU)
// W row-major (NOUT,K). One wave = 16-row M tile, all N columns.
// A/B fragments loaded as b128; f32->f16 packed on the fly;
// accumulate f32 via v_wmma_f32_16x16x32_f16.
// ---------------------------------------------------------------------------
template <int K, int NOUT, int ACT, int RES>
__global__ void gemm_wmma_kernel(const float* __restrict__ A,
                                 const float* __restrict__ W,
                                 const float* __restrict__ bias,
                                 const float* __restrict__ res,
                                 float* __restrict__ out, int M) {
  const int wave = (blockIdx.x * blockDim.x + threadIdx.x) >> 5;
  const int lane = threadIdx.x & 31;
  const int row0 = wave * 16;
  if (row0 >= M) return;                  // wave-uniform; EXEC stays all-ones
  const int rl = lane & 15;
  const int hi = lane >> 4;               // 0 / 1
  constexpr int NT = NOUT / 16;
  constexpr int KC = K / 32;

  v8f acc[NT];
  const v8f vzero = {0.f, 0.f, 0.f, 0.f, 0.f, 0.f, 0.f, 0.f};
  #pragma unroll
  for (int t = 0; t < NT; ++t) acc[t] = vzero;

  #pragma unroll
  for (int kc = 0; kc < KC; ++kc) {
    // ----- A fragment (ISA 7.12.2, 16-bit A 16x32):
    // lane row = rl; this lane needs K = {kb..kb+7, 16+kb..16+kb+7}, kb=8*hi.
    // Both runs are 8 contiguous, 32B-aligned floats -> 4x global_load_b128.
    const float* arow = A + (size_t)(row0 + rl) * K + kc * 32;
    const int kb = hi * 8;
    const v4f* alo = reinterpret_cast<const v4f*>(arow + kb);
    const v4f* ahi = reinterpret_cast<const v4f*>(arow + 16 + kb);
    v4f a0 = alo[0], a1 = alo[1], a2 = ahi[0], a3 = ahi[1];
    v16h af;
    #pragma unroll
    for (int e = 0; e < 4; ++e) {
      af[e]      = (_Float16)a0[e];
      af[4 + e]  = (_Float16)a1[e];
      af[8 + e]  = (_Float16)a2[e];
      af[12 + e] = (_Float16)a3[e];
    }

    #pragma unroll
    for (int t = 0; t < NT; ++t) {
      // ----- B fragment: col = rl; K = {kb..kb+7, 16+kb..16+kb+7}, same as A
      const int col = t * 16 + rl;
      const v4f* w4 = reinterpret_cast<const v4f*>(W + (size_t)col * K + kc * 32 + kb);
      v4f b0 = w4[0], b1 = w4[1], b2 = w4[4], b3 = w4[5];
      v16h bf;
      #pragma unroll
      for (int e = 0; e < 4; ++e) {
        bf[e]      = (_Float16)b0[e];
        bf[4 + e]  = (_Float16)b1[e];
        bf[8 + e]  = (_Float16)b2[e];
        bf[12 + e] = (_Float16)b3[e];
      }
      acc[t] = __builtin_amdgcn_wmma_f32_16x16x32_f16(
          false, af, false, bf, (short)0, acc[t], false, false);
    }
  }

  // ----- epilogue: C/D layout -> M = v + 8*hi, N = lane&15
  #pragma unroll
  for (int t = 0; t < NT; ++t) {
    const int col = t * 16 + rl;
    #pragma unroll
    for (int v = 0; v < 8; ++v) {
      const int row = row0 + v + hi * 8;
      float val = acc[t][v] + bias[col];
      if (RES) val += res[(size_t)row * NOUT + col];
      if (ACT == 1) val = 0.5f * val * (1.0f + erff(val * 0.70710678118f));
      out[(size_t)row * NOUT + col] = val;
    }
  }
}

// ---------------------------------------------------------------------------
// Dynamic position bias MLP + gather into bias table [HEADS][N][N]
// ---------------------------------------------------------------------------
static __device__ __forceinline__
void dpb_ln_relu(const float* in, float* out, const float* g, const float* b) {
  float mu = 0.f;
  #pragma unroll
  for (int i = 0; i < PD; ++i) mu += in[i];
  mu *= (1.0f / PD);
  float var = 0.f;
  #pragma unroll
  for (int i = 0; i < PD; ++i) { float d = in[i] - mu; var += d * d; }
  var *= (1.0f / PD);
  float inv = rsqrtf(var + 1e-5f);
  #pragma unroll
  for (int i = 0; i < PD; ++i) {
    float t = (in[i] - mu) * inv * g[i] + b[i];
    out[i] = t > 0.f ? t : 0.f;
  }
}

__global__ void dpb_kernel(const float* pp_w, const float* pp_b,
                           const float* l1g, const float* l1b, const float* w1, const float* b1,
                           const float* l2g, const float* l2b, const float* w2, const float* b2,
                           const float* l3g, const float* l3b, const float* w3, const float* b3,
                           float* __restrict__ pos2, float* __restrict__ bias, int G) {
  const int D = 2 * G - 1;
  const int P = D * D;
  const int N = G * G;
  for (int t = threadIdx.x; t < P; t += blockDim.x) {
    float r0 = (float)(t / D - (G - 1));
    float r1 = (float)(t % D - (G - 1));
    float v[PD], u[PD];
    #pragma unroll
    for (int i = 0; i < PD; ++i) v[i] = r0 * pp_w[i * 2 + 0] + r1 * pp_w[i * 2 + 1] + pp_b[i];
    dpb_ln_relu(v, u, l1g, l1b);
    #pragma unroll
    for (int i = 0; i < PD; ++i) {
      float s = b1[i];
      #pragma unroll
      for (int j = 0; j < PD; ++j) s += u[j] * w1[i * PD + j];
      v[i] = s;
    }
    dpb_ln_relu(v, u, l2g, l2b);
    #pragma unroll
    for (int i = 0; i < PD; ++i) {
      float s = b2[i];
      #pragma unroll
      for (int j = 0; j < PD; ++j) s += u[j] * w2[i * PD + j];
      v[i] = s;
    }
    dpb_ln_relu(v, u, l3g, l3b);
    for (int h = 0; h < HEADS; ++h) {
      float s = b3[h];
      #pragma unroll
      for (int j = 0; j < PD; ++j) s += u[j] * w3[h * PD + j];
      pos2[t * HEADS + h] = s;
    }
  }
  __syncthreads();
  for (int e = threadIdx.x; e < HEADS * N * N; e += blockDim.x) {
    int m  = e % N;
    int n  = (e / N) % N;
    int h  = e / (N * N);
    int nh = n / G, nw = n % G, mh = m / G, mw = m % G;
    int idx = (nh - mh + G - 1) * D + (nw - mw + G - 1);
    bias[e] = pos2[idx * HEADS + h];
  }
}

// ---------------------------------------------------------------------------
// Attention, lsda_flag==0, G=5, N=25. One wave per (group, head).
// ---------------------------------------------------------------------------
__global__ void attn5_kernel(const float* __restrict__ qkv,
                             const float* __restrict__ bias,
                             float* __restrict__ attnout) {
  __shared__ float skv[8][2][25][HD];      // per-wave K and V stage
  const int wv   = threadIdx.x >> 5;
  const int lane = threadIdx.x & 31;
  const int gid  = blockIdx.x * 8 + wv;    // (group, head) id
  const int head = gid & 1;
  const int g    = gid >> 1;
  const int b    = g / (64 * 64);
  const int gr   = g % (64 * 64);
  const int gh   = gr / 64, gw = gr % 64;

  if (lane < 25) {
    int nh = lane / 5, nw = lane % 5;
    size_t tok = (size_t)b * LTOK + (size_t)(gh * 5 + nh) * SZ + (gw * 5 + nw);
    const v4f* kb4 = reinterpret_cast<const v4f*>(qkv + tok * 96 + 32 + head * HD);
    const v4f* vb4 = reinterpret_cast<const v4f*>(qkv + tok * 96 + 64 + head * HD);
    #pragma unroll
    for (int i = 0; i < HD / 4; ++i) {
      v4f kk = kb4[i], vv = vb4[i];
      #pragma unroll
      for (int e = 0; e < 4; ++e) {
        skv[wv][0][lane][i * 4 + e] = kk[e];
        skv[wv][1][lane][i * 4 + e] = vv[e];
      }
    }
  }
  __syncthreads();
  if (lane >= 25) return;

  const int n  = lane;
  const int nh = n / 5, nw = n % 5;
  size_t tok = (size_t)b * LTOK + (size_t)(gh * 5 + nh) * SZ + (gw * 5 + nw);
  const v4f* qb4 = reinterpret_cast<const v4f*>(qkv + tok * 96 + head * HD);
  float q[HD];
  #pragma unroll
  for (int i = 0; i < HD / 4; ++i) {
    v4f qq = qb4[i];
    #pragma unroll
    for (int e = 0; e < 4; ++e) q[i * 4 + e] = qq[e] * 0.25f;   // hd^-0.5
  }

  float s[25];
  float mx = -1e30f;
  for (int m = 0; m < 25; ++m) {
    float a = bias[head * 625 + n * 25 + m];
    #pragma unroll
    for (int d = 0; d < HD; ++d) a += q[d] * skv[wv][0][m][d];
    s[m] = a;
    mx = fmaxf(mx, a);
  }
  float sum = 0.f;
  for (int m = 0; m < 25; ++m) { s[m] = expf(s[m] - mx); sum += s[m]; }
  float inv = 1.0f / sum;

  v4f* op4 = reinterpret_cast<v4f*>(attnout + tok * CB + head * HD);
  #pragma unroll
  for (int i = 0; i < HD / 4; ++i) {
    v4f o;
    #pragma unroll
    for (int e = 0; e < 4; ++e) {
      float a = 0.f;
      for (int m = 0; m < 25; ++m) a += s[m] * skv[wv][1][m][i * 4 + e];
      o[e] = a * inv;
    }
    op4[i] = o;
  }
}

// ---------------------------------------------------------------------------
// Attention, lsda_flag==1, I=160, Gh=Gw=2, N=4. One thread per (group, head).
// ---------------------------------------------------------------------------
__global__ void attn4_kernel(const float* __restrict__ qkv,
                             const float* __restrict__ bias,
                             float* __restrict__ attnout) {
  int id = blockIdx.x * blockDim.x + threadIdx.x;
  int total = BATCH * HS * HS * HEADS;
  if (id >= total) return;
  int head = id & 1;
  int g    = id >> 1;
  int b    = g / (HS * HS);
  int gr   = g % (HS * HS);
  int i1   = gr / HS, i2 = gr % HS;

  size_t tok[4];
  #pragma unroll
  for (int n = 0; n < 4; ++n) {
    int g1 = n >> 1, g2 = n & 1;
    tok[n] = (size_t)b * LTOK + (size_t)(g1 * HS + i1) * SZ + (g2 * HS + i2);
  }
  #pragma unroll
  for (int n = 0; n < 4; ++n) {
    const v4f* qb4 = reinterpret_cast<const v4f*>(qkv + tok[n] * 96 + head * HD);
    float q[HD];
    #pragma unroll
    for (int i = 0; i < HD / 4; ++i) {
      v4f qq = qb4[i];
      #pragma unroll
      for (int e = 0; e < 4; ++e) q[i * 4 + e] = qq[e] * 0.25f;
    }
    float s[4];
    float mx = -1e30f;
    #pragma unroll
    for (int m = 0; m < 4; ++m) {
      const v4f* kb4 = reinterpret_cast<const v4f*>(qkv + tok[m] * 96 + 32 + head * HD);
      float a = bias[head * 16 + n * 4 + m];
      #pragma unroll
      for (int i = 0; i < HD / 4; ++i) {
        v4f kk = kb4[i];
        #pragma unroll
        for (int e = 0; e < 4; ++e) a += q[i * 4 + e] * kk[e];
      }
      s[m] = a;
      mx = fmaxf(mx, a);
    }
    float sum = 0.f;
    #pragma unroll
    for (int m = 0; m < 4; ++m) { s[m] = expf(s[m] - mx); sum += s[m]; }
    float inv = 1.0f / sum;
    float o[HD];
    #pragma unroll
    for (int d = 0; d < HD; ++d) o[d] = 0.f;
    #pragma unroll
    for (int m = 0; m < 4; ++m) {
      const v4f* vb4 = reinterpret_cast<const v4f*>(qkv + tok[m] * 96 + 64 + head * HD);
      float p = s[m] * inv;
      #pragma unroll
      for (int i = 0; i < HD / 4; ++i) {
        v4f vv = vb4[i];
        #pragma unroll
        for (int e = 0; e < 4; ++e) o[i * 4 + e] += p * vv[e];
      }
    }
    v4f* op4 = reinterpret_cast<v4f*>(attnout + tok[n] * CB + head * HD);
    #pragma unroll
    for (int i = 0; i < HD / 4; ++i) {
      v4f ov;
      #pragma unroll
      for (int e = 0; e < 4; ++e) ov[e] = o[i * 4 + e];
      op4[i] = ov;
    }
  }
}

// ---------------------------------------------------------------------------
// IDWT: ll from transformer output, high bands recomputed from original x
// ---------------------------------------------------------------------------
__global__ void idwt_kernel(const float* __restrict__ x,
                            const float* __restrict__ f1,
                            const float* __restrict__ f2,
                            float* __restrict__ transy) {
  int idx = blockIdx.x * blockDim.x + threadIdx.x;
  int total = BATCH * C_IN * HS * HS;
  if (idx >= total) return;
  int j = idx % HS;  int t = idx / HS;
  int i = t % HS;    t /= HS;
  int ch = t % C_IN; int b = t / C_IN;

  // ll = tf2[b, ch, i, j]
  const float* fh = (ch < CB) ? f1 : f2;
  int cc = ch % CB;
  size_t flat = (size_t)cc * (SZ * SZ) + (size_t)i * SZ + j;
  size_t l = flat / CB;
  int    c = (int)(flat % CB);
  float ll = fh[((size_t)b * LTOK + l) * CB + c];

  const float* xp = x + (((size_t)b * C_IN + ch) * SZ + 2 * i) * SZ + 2 * j;
  float a0 = xp[0], b0 = xp[1], c0 = xp[SZ], d0 = xp[SZ + 1];
  float lh = 0.5f * (-a0 - b0 + c0 + d0);
  float hl = 0.5f * (-a0 + b0 - c0 + d0);
  float hh = 0.5f * ( a0 - b0 - c0 + d0);

  float av = (ll - lh - hl + hh) * 0.5f;
  float bv = (ll - lh + hl - hh) * 0.5f;
  float cv = (ll + lh - hl - hh) * 0.5f;
  float dv = (ll + lh + hl + hh) * 0.5f;

  float* yp = transy + (((size_t)b * C_IN + ch) * SZ + 2 * i) * SZ + 2 * j;
  yp[0] = av; yp[1] = bv; yp[SZ] = cv; yp[SZ + 1] = dv;
}

// ---------------------------------------------------------------------------
// Grouped 3x3 conv, feature_group_count=4, SAME padding
// ---------------------------------------------------------------------------
__global__ void conv_kernel(const float* __restrict__ x, const float* __restrict__ w,
                            const float* __restrict__ bias, float* __restrict__ y) {
  size_t idx = (size_t)blockIdx.x * blockDim.x + threadIdx.x;
  size_t total = (size_t)BATCH * C_IN * SZ * SZ;
  if (idx >= total) return;
  int wq = (int)(idx % SZ);  size_t t = idx / SZ;
  int h  = (int)(t % SZ);    t /= SZ;
  int o  = (int)(t % C_IN);  int b = (int)(t / C_IN);
  int gbase = (o / 16) * 16;
  float acc = bias[o];
  for (int i = 0; i < 16; ++i) {
    const float* xp = x + ((size_t)b * C_IN + gbase + i) * (SZ * SZ);
    const float* wp = w + ((size_t)o * 16 + i) * 9;
    #pragma unroll
    for (int kh = 0; kh < 3; ++kh) {
      int hy = h + kh - 1;
      if (hy < 0 || hy >= SZ) continue;
      #pragma unroll
      for (int kw = 0; kw < 3; ++kw) {
        int wx = wq + kw - 1;
        if (wx < 0 || wx >= SZ) continue;
        acc += xp[(size_t)hy * SZ + wx] * wp[kh * 3 + kw];
      }
    }
  }
  y[idx] = acc;
}

// ---------------------------------------------------------------------------
// Channel select: pool -> scale -> linear -> softmax -> sigmoid -> pick
// ---------------------------------------------------------------------------
__global__ void pool_kernel(const float* __restrict__ transy,
                            const float* __restrict__ convsy,
                            float* __restrict__ pool) {
  __shared__ float red[256];
  int bc = blockIdx.x;                  // B * 128 blocks
  int b  = bc / (2 * C_IN);
  int ch = bc % (2 * C_IN);
  const float* src = (ch < C_IN)
      ? transy + ((size_t)b * C_IN + ch) * (SZ * SZ)
      : convsy + ((size_t)b * C_IN + (ch - C_IN)) * (SZ * SZ);
  float s = 0.f;
  for (int i = threadIdx.x; i < SZ * SZ; i += blockDim.x) s += src[i];
  red[threadIdx.x] = s;
  __syncthreads();
  for (int st = 128; st > 0; st >>= 1) {
    if ((int)threadIdx.x < st) red[threadIdx.x] += red[threadIdx.x + st];
    __syncthreads();
  }
  if (threadIdx.x == 0) pool[bc] = red[0] * (1.0f / (SZ * SZ));
}

__global__ void ca_kernel(const float* __restrict__ pool,
                          const float* __restrict__ dw_w, const float* __restrict__ dw_b,
                          const float* __restrict__ pw_w, const float* __restrict__ pw_b,
                          float* __restrict__ ca) {
  __shared__ float pp[128], red[128];
  int t = threadIdx.x;                  // 128 threads
  for (int b = 0; b < BATCH; ++b) {
    pp[t] = pool[b * 128 + t] * dw_w[t] + dw_b[t];
    __syncthreads();
    float z = pw_b[t];
    for (int k = 0; k < 128; ++k) z += pw_w[t * 128 + k] * pp[k];
    red[t] = z;
    __syncthreads();
    for (int st = 64; st > 0; st >>= 1) {
      if (t < st) red[t] = fmaxf(red[t], red[t + st]);
      __syncthreads();
    }
    float mx = red[0];
    __syncthreads();
    float e = expf(z - mx);
    red[t] = e;
    __syncthreads();
    for (int st = 64; st > 0; st >>= 1) {
      if (t < st) red[t] += red[t + st];
      __syncthreads();
    }
    float sm = e / red[0];
    ca[b * 128 + t] = 1.0f / (1.0f + expf(-sm));
    __syncthreads();
  }
}

__global__ void select_kernel(const float* __restrict__ tx, const float* __restrict__ cy,
                              const float* __restrict__ ca, float* __restrict__ out) {
  size_t idx = (size_t)blockIdx.x * blockDim.x + threadIdx.x;
  size_t total = (size_t)BATCH * C_IN * SZ * SZ;
  if (idx >= total) return;
  int ch = (int)((idx / (SZ * SZ)) % C_IN);
  int b  = (int)(idx / ((size_t)C_IN * SZ * SZ));
  float cax = ca[b * 128 + ch];
  float cay = ca[b * 128 + C_IN + ch];
  float xv = tx[idx], yv = cy[idx];
  out[idx] = (cax > cay) ? xv : ((cax < cay) ? yv : 0.5f * (xv + yv));
}

// ---------------------------------------------------------------------------
// Host-side block driver: one transformer block over f (M = 204800 rows)
// ---------------------------------------------------------------------------
static void run_block(float* f, void* const* P, const float* bias, int flag,
                      float* xln, float* big, float* att, hipStream_t stream) {
  const float* n1g    = (const float*)P[0];
  const float* n1b    = (const float*)P[1];
  const float* qkv_w  = (const float*)P[2];
  const float* qkv_b  = (const float*)P[3];
  const float* proj_w = (const float*)P[4];
  const float* proj_b = (const float*)P[5];
  const float* n2g    = (const float*)P[6];
  const float* n2b    = (const float*)P[7];
  const float* fc1_w  = (const float*)P[8];
  const float* fc1_b  = (const float*)P[9];
  const float* fc2_w  = (const float*)P[10];
  const float* fc2_b  = (const float*)P[11];

  const int gemmBlocks = MTOK / 128;   // 8 waves/block * 16 rows/wave

  ln_kernel<<<cdiv(MTOK, 256), 256, 0, stream>>>(f, n1g, n1b, xln, MTOK);
  gemm_wmma_kernel<32, 96, 0, 0><<<gemmBlocks, 256, 0, stream>>>(
      xln, qkv_w, qkv_b, nullptr, big, MTOK);
  if (flag == 0)
    attn5_kernel<<<BATCH * 64 * 64 * HEADS / 8, 256, 0, stream>>>(big, bias, att);
  else
    attn4_kernel<<<cdiv(BATCH * HS * HS * HEADS, 256), 256, 0, stream>>>(big, bias, att);
  gemm_wmma_kernel<32, 32, 0, 1><<<gemmBlocks, 256, 0, stream>>>(
      att, proj_w, proj_b, f, f, MTOK);
  ln_kernel<<<cdiv(MTOK, 256), 256, 0, stream>>>(f, n2g, n2b, xln, MTOK);
  gemm_wmma_kernel<32, 128, 1, 0><<<gemmBlocks, 256, 0, stream>>>(
      xln, fc1_w, fc1_b, nullptr, big, MTOK);
  gemm_wmma_kernel<128, 32, 0, 1><<<gemmBlocks, 256, 0, stream>>>(
      big, fc2_w, fc2_b, f, f, MTOK);
}

extern "C" void kernel_launch(void* const* d_in, const int* in_sizes, int n_in,
                              void* d_out, int out_size, void* d_ws, size_t ws_size,
                              hipStream_t stream) {
  (void)in_sizes; (void)n_in; (void)out_size; (void)ws_size;

  const float* x = (const float*)d_in[0];
  void* const* P1 = d_in + 1;           // 26 tensors of t1
  void* const* P2 = d_in + 27;          // 26 tensors of t2
  const float* conv_w = (const float*)d_in[53];
  const float* conv_b = (const float*)d_in[54];
  const float* dw_w   = (const float*)d_in[55];
  const float* dw_b   = (const float*)d_in[56];
  const float* pw_w   = (const float*)d_in[57];
  const float* pw_b   = (const float*)d_in[58];

  float* ws = (float*)d_ws;
  const size_t S32  = (size_t)MTOK * CB;               //  6,553,600 floats
  const size_t SBIG = (size_t)MTOK * MLPD;             // 26,214,400 floats
  const size_t SIMG = (size_t)BATCH * C_IN * SZ * SZ;  // 13,107,200 floats

  float* f1     = ws;                 // (M, 32)
  float* f2     = f1 + S32;
  float* xln    = f2 + S32;
  float* att    = xln + S32;
  float* big    = att + S32;          // qkv (M,96) / fc1 out (M,128)
  float* transy = big + SBIG;
  float* convsy = transy + SIMG;
  float* biasT  = convsy + SIMG;      // [2][N][N] max 1250
  float* posT   = biasT + 1280;       // dpb scratch, max 162

  // 1) DWT + re-pack into the two 32-channel halves
  dwt_feat_kernel<<<cdiv(BATCH * C_IN * HS * HS, 256), 256, 0, stream>>>(x, f1, f2);

  // 2) t1 blocks (G=5 grouped attention) on both halves
  dpb_kernel<<<1, 128, 0, stream>>>(
      (const float*)P1[12], (const float*)P1[13],
      (const float*)P1[14], (const float*)P1[15], (const float*)P1[16], (const float*)P1[17],
      (const float*)P1[18], (const float*)P1[19], (const float*)P1[20], (const float*)P1[21],
      (const float*)P1[22], (const float*)P1[23], (const float*)P1[24], (const float*)P1[25],
      posT, biasT, 5);
  run_block(f1, P1, biasT, 0, xln, big, att, stream);
  run_block(f2, P1, biasT, 0, xln, big, att, stream);

  // 3) t2 blocks (I=160 long-range attention, N=4) on both halves
  dpb_kernel<<<1, 128, 0, stream>>>(
      (const float*)P2[12], (const float*)P2[13],
      (const float*)P2[14], (const float*)P2[15], (const float*)P2[16], (const float*)P2[17],
      (const float*)P2[18], (const float*)P2[19], (const float*)P2[20], (const float*)P2[21],
      (const float*)P2[22], (const float*)P2[23], (const float*)P2[24], (const float*)P2[25],
      posT, biasT, 2);
  run_block(f1, P2, biasT, 1, xln, big, att, stream);
  run_block(f2, P2, biasT, 1, xln, big, att, stream);

  // 4) IDWT branch + conv branch
  idwt_kernel<<<cdiv(BATCH * C_IN * HS * HS, 256), 256, 0, stream>>>(x, f1, f2, transy);
  conv_kernel<<<(int)cdiv((int)SIMG, 256), 256, 0, stream>>>(x, conv_w, conv_b, convsy);

  // 5) channel selection
  float* poolB = posT + 256;
  pool_kernel<<<BATCH * 2 * C_IN, 256, 0, stream>>>(transy, convsy, poolB);
  float* caB = poolB + 256;
  ca_kernel<<<1, 128, 0, stream>>>(poolB, dw_w, dw_b, pw_w, pw_b, caB);
  select_kernel<<<(int)cdiv((int)SIMG, 256), 256, 0, stream>>>(
      transy, convsy, caB, (float*)d_out);
}
